// RNNDecoder_317827580214
// MI455X (gfx1250) — matrix operand order repaired and checked by reading, not compiled, not run on hardware
//
#include <hip/hip_runtime.h>
#include <hip/hip_bf16.h>
#include <math.h>

// ---------------------------------------------------------------------------
// RNN decoder (2-layer LSTM + generator + log_softmax), T=128 B=32 E=H=512
// G=2048 V=32000.  bf16 WMMA (v_wmma_f32_16x16x32_bf16) for all GEMMs.
// Weights are packed ONCE per launch into the exact B-fragment layout so the
// GEMM inner loop is 8 contiguous dword loads per lane + one WMMA per K-tile.
// All weights (~43 MB packed) are L2-resident on MI455X (192 MB L2).
// ---------------------------------------------------------------------------

#define TT 128
#define BB 32
#define EE 512
#define HH 512
#define GG 2048
#define VV 32000
#define KP0 1568   // 1537 padded to 49*32
#define KT0 49
#define KP1 1024   // 32 k-tiles
#define KT1 32
#define KPG 512    // 16 k-tiles
#define KTG 16

typedef __attribute__((ext_vector_type(16))) __bf16 v16bf;
typedef __attribute__((ext_vector_type(8)))  float  v8f;

union Frag { unsigned int u[8]; v16bf v; };

__device__ __forceinline__ unsigned short f32_to_bf16(float f) {
  unsigned int u = __float_as_uint(f);
  unsigned int r = u + 0x7FFFu + ((u >> 16) & 1u);   // round-to-nearest-even
  return (unsigned short)(r >> 16);
}

// ---------------------------------------------------------------------------
// Pack W = [Wih | Whh] (each stored [G, K] row-major, gemm is x @ W.T) into
// B-matrix 32x16 bf16 fragments:
//   lanes 0-15: col n = lane, K = 0..15  (VGPR v -> K = 2v, 2v+1)
//   lanes 16-31: col n = lane-16, K = 16..31
// frag index: ((ntile*ktiles + ktile)*32 + lane)*8 + v
// ---------------------------------------------------------------------------
__global__ void pack_weights_kernel(const float* __restrict__ Wih, int Kih,
                                    const float* __restrict__ Whh, int Khh,
                                    unsigned int* __restrict__ frag, int ktiles) {
  int ntile = blockIdx.x, ktile = blockIdx.y, lane = threadIdx.x;
  int g    = ntile * 16 + (lane & 15);
  int koff = (lane >> 4) ? 16 : 0;
  unsigned int* dst = frag + ((size_t)(ntile * ktiles + ktile) * 32 + lane) * 8;
#pragma unroll
  for (int v = 0; v < 8; ++v) {
    int k0 = ktile * 32 + koff + 2 * v;
    float f0 = 0.f, f1 = 0.f;
    if (k0 < Kih)                     f0 = Wih[(size_t)g * Kih + k0];
    else if (Whh && k0 < Kih + Khh)   f0 = Whh[(size_t)g * Khh + (k0 - Kih)];
    int k1 = k0 + 1;
    if (k1 < Kih)                     f1 = Wih[(size_t)g * Kih + k1];
    else if (Whh && k1 < Kih + Khh)   f1 = Whh[(size_t)g * Khh + (k1 - Kih)];
    dst[v] = ((unsigned int)f32_to_bf16(f1) << 16) | (unsigned int)f32_to_bf16(f0);
  }
}

// ---------------------------------------------------------------------------
// WMMA GEMM: out[M=32, N] = A[32, Kpad](bf16 row-major) * Wfrag + bias0+bias1
// One wave (32 threads) per N-tile; computes BOTH M-tiles so each B fragment
// is loaded once.  A-fragment layout (16x32 bf16, same K order as the B packing):
//   lanes 0-15: row M=lane, K = 0..15 (VGPR v -> K = 2v, 2v+1)
//   lanes 16-31: row M=lane-16, K = 16..31.
// ---------------------------------------------------------------------------
__global__ void wmma_gemm_bias_kernel(const unsigned short* __restrict__ abf, int Kpad,
                                      const unsigned int* __restrict__ wfrag, int ktiles,
                                      const float* __restrict__ bias0,
                                      const float* __restrict__ bias1,
                                      float* __restrict__ outp, int N) {
  const int ntile = blockIdx.x;
  const int lane  = threadIdx.x;
  v8f acc0 = {}, acc1 = {};
  const int mlo = lane & 15;
  const int ah8 = ((lane >> 4) & 1) * 16;         // A K-offset per lane half
  const unsigned short* arow0 = abf + (size_t)mlo * Kpad;
  const unsigned short* arow1 = abf + (size_t)(16 + mlo) * Kpad;
  const unsigned int* wp = wfrag + ((size_t)ntile * ktiles * 32 + lane) * 8;

  for (int kt = 0; kt < ktiles; ++kt) {
    Frag b, a0, a1;
#pragma unroll
    for (int v = 0; v < 8; ++v) b.u[v] = wp[v];
    __builtin_prefetch(wp + 256, 0, 0);           // global_prefetch_b8 (next tile)
    wp += 256;
    const int kbase = kt * 32 + ah8;
#pragma unroll
    for (int v = 0; v < 8; ++v) {
      int k = kbase + 2 * v;
      a0.u[v] = *(const unsigned int*)(arow0 + k);
      a1.u[v] = *(const unsigned int*)(arow1 + k);
    }
    acc0 = __builtin_amdgcn_wmma_f32_16x16x32_bf16(false, a0.v, false, b.v,
                                                   (short)0, acc0, false, false);
    acc1 = __builtin_amdgcn_wmma_f32_16x16x32_bf16(false, a1.v, false, b.v,
                                                   (short)0, acc1, false, false);
  }
  const int n = ntile * 16 + (lane & 15);
  float badd = (bias0 ? bias0[n] : 0.f) + (bias1 ? bias1[n] : 0.f);
  const int mb = (lane >> 4) ? 8 : 0;             // C/D: lanes 16-31 hold M=8..15
#pragma unroll
  for (int r = 0; r < 8; ++r) {
    outp[(size_t)(mb + r) * N + n]      = acc0[r] + badd;
    outp[(size_t)(16 + mb + r) * N + n] = acc1[r] + badd;
  }
}

// ---------------------------------------------------------------------------
// Build layer-0 input  xh0[b, 0..1567] (bf16):
//   [ word_emb+special (512) | countdown (1) | out_prev (512) | h0_prev (512) | 0-pad ]
// ---------------------------------------------------------------------------
__global__ void build_xh0_kernel(const int* __restrict__ ids,
                                 const float* __restrict__ cdown,
                                 const float* __restrict__ word_emb,
                                 const float* __restrict__ special_emb,
                                 const float* __restrict__ out_prev,
                                 const float* __restrict__ h0_prev,
                                 unsigned short* __restrict__ xh0, int t) {
  int idx = blockIdx.x * blockDim.x + threadIdx.x;
  if (idx >= BB * KP0) return;
  int b = idx / KP0, j = idx % KP0;
  float val;
  if (j < EE) {
    int id = ids[t * BB + b];
    int sp = (id < 3) ? (id + 1) : 0;
    val = word_emb[(size_t)id * EE + j] + special_emb[sp * EE + j];
  } else if (j == EE) {
    val = cdown[t * BB + b];
  } else if (j < EE + 1 + HH) {
    val = out_prev[b * HH + (j - EE - 1)];
  } else if (j < EE + 1 + 2 * HH) {
    val = h0_prev[b * HH + (j - EE - 1 - HH)];
  } else {
    val = 0.f;
  }
  xh0[idx] = f32_to_bf16(val);
}

// ---------------------------------------------------------------------------
// LSTM pointwise epilogue (torch gate order i,f,g,o along G).
// For layer 0: also emits xh1 = [ h0_new (bf16) | h1_old (bf16) ].
// For layer 1: also emits out (f32) and out (bf16) for the generator A matrix.
// ---------------------------------------------------------------------------
__global__ void lstm_epilogue_kernel(const float* __restrict__ gates,
                                     float* __restrict__ h_l, float* __restrict__ c_l,
                                     unsigned short* __restrict__ xh_next,
                                     const float* __restrict__ h_other,
                                     float* __restrict__ out_f32,
                                     unsigned short* __restrict__ out_bf) {
  int idx = blockIdx.x * blockDim.x + threadIdx.x;
  if (idx >= BB * HH) return;
  int b = idx / HH, j = idx % HH;
  float gi = gates[(size_t)b * GG + j];
  float gf = gates[(size_t)b * GG + HH + j];
  float gg = gates[(size_t)b * GG + 2 * HH + j];
  float go = gates[(size_t)b * GG + 3 * HH + j];
  float si = 1.f / (1.f + __expf(-gi));
  float sf = 1.f / (1.f + __expf(-gf));
  float so = 1.f / (1.f + __expf(-go));
  float c_new = sf * c_l[idx] + si * tanhf(gg);
  float h_new = so * tanhf(c_new);
  c_l[idx] = c_new;
  h_l[idx] = h_new;
  if (xh_next) {
    xh_next[(size_t)b * KP1 + j]      = f32_to_bf16(h_new);
    xh_next[(size_t)b * KP1 + HH + j] = f32_to_bf16(h_other[idx]);
  }
  if (out_f32) {
    out_f32[idx] = h_new;
    out_bf[idx]  = f32_to_bf16(h_new);
  }
}

// ---------------------------------------------------------------------------
// Row-wise log_softmax over V=32000: one block (8 waves) per batch row.
// ---------------------------------------------------------------------------
__global__ void logsoftmax_kernel(const float* __restrict__ logits,
                                  float* __restrict__ scores_out) {
  __shared__ float red[256];
  const int b = blockIdx.x, tid = threadIdx.x;
  const float* row = logits + (size_t)b * VV;
  float m = -INFINITY;
  for (int v = tid; v < VV; v += 256) m = fmaxf(m, row[v]);
  red[tid] = m; __syncthreads();
  for (int s = 128; s > 0; s >>= 1) {
    if (tid < s) red[tid] = fmaxf(red[tid], red[tid + s]);
    __syncthreads();
  }
  m = red[0]; __syncthreads();
  float sum = 0.f;
  for (int v = tid; v < VV; v += 256) sum += __expf(row[v] - m);
  red[tid] = sum; __syncthreads();
  for (int s = 128; s > 0; s >>= 1) {
    if (tid < s) red[tid] += red[tid + s];
    __syncthreads();
  }
  float lse = m + __logf(red[0]);
  float* dst = scores_out + (size_t)b * VV;
  for (int v = tid; v < VV; v += 256) dst[v] = row[v] - lse;
}

// --------------------------- state init / finalize -------------------------
__global__ void init_state_kernel(const float* __restrict__ h0,
                                  const float* __restrict__ c0,
                                  const float* __restrict__ prev_out,
                                  float* __restrict__ h_state,
                                  float* __restrict__ c_state,
                                  float* __restrict__ out_f32) {
  int idx = blockIdx.x * blockDim.x + threadIdx.x;
  const int SH = 2 * BB * HH;
  if (idx < SH)      h_state[idx] = h0[idx];
  if (idx < SH)      c_state[idx] = c0[idx];
  if (idx < BB * HH) out_f32[idx] = prev_out[idx];
}

__global__ void finalize_kernel(const float* __restrict__ h_state,
                                const float* __restrict__ c_state,
                                const float* __restrict__ out_f32,
                                float* __restrict__ tail) {
  int idx = blockIdx.x * blockDim.x + threadIdx.x;
  const int SH = 2 * BB * HH;
  if (idx < SH)                    tail[idx] = h_state[idx];
  else if (idx < 2 * SH)           tail[idx] = c_state[idx - SH];
  else if (idx < 2 * SH + BB * HH) tail[idx] = out_f32[idx - 2 * SH];
}

// ---------------------------------------------------------------------------
extern "C" void kernel_launch(void* const* d_in, const int* in_sizes, int n_in,
                              void* d_out, int out_size, void* d_ws, size_t ws_size,
                              hipStream_t stream) {
  (void)in_sizes; (void)n_in; (void)out_size; (void)ws_size;
  const int*   ids      = (const int*)  d_in[0];
  const float* cdown    = (const float*)d_in[1];
  const float* word_emb = (const float*)d_in[2];
  const float* spec_emb = (const float*)d_in[3];
  const float* h0       = (const float*)d_in[4];
  const float* c0       = (const float*)d_in[5];
  const float* prev_out = (const float*)d_in[6];
  const float* Wih0     = (const float*)d_in[7];
  const float* Whh0     = (const float*)d_in[8];
  const float* bih0     = (const float*)d_in[9];
  const float* bhh0     = (const float*)d_in[10];
  const float* Wih1     = (const float*)d_in[11];
  const float* Whh1     = (const float*)d_in[12];
  const float* bih1     = (const float*)d_in[13];
  const float* bhh1     = (const float*)d_in[14];
  const float* Wg       = (const float*)d_in[15];
  const float* bg       = (const float*)d_in[16];
  float* out_f = (float*)d_out;

  // ---- workspace carve-up (256B aligned), total ~48.5 MB ----
  char* ws = (char*)d_ws;
  size_t off = 0;
  auto carve = [&](size_t bytes) { char* p = ws + off; off = (off + bytes + 255) & ~(size_t)255; return p; };
  unsigned int*  w0frag  = (unsigned int*) carve((size_t)(GG/16) * KT0 * 32 * 8 * 4);  // 6.4 MB
  unsigned int*  w1frag  = (unsigned int*) carve((size_t)(GG/16) * KT1 * 32 * 8 * 4);  // 4.2 MB
  unsigned int*  wgfrag  = (unsigned int*) carve((size_t)(VV/16) * KTG * 32 * 8 * 4);  // 32.8 MB
  unsigned short* xh0    = (unsigned short*)carve((size_t)BB * KP0 * 2);
  unsigned short* xh1    = (unsigned short*)carve((size_t)BB * KP1 * 2);
  unsigned short* outbf  = (unsigned short*)carve((size_t)BB * HH * 2);
  float* gates0  = (float*)carve((size_t)BB * GG * 4);
  float* gates1  = (float*)carve((size_t)BB * GG * 4);
  float* logits  = (float*)carve((size_t)BB * VV * 4);
  float* h_state = (float*)carve((size_t)2 * BB * HH * 4);
  float* c_state = (float*)carve((size_t)2 * BB * HH * 4);
  float* outf32  = (float*)carve((size_t)BB * HH * 4);

  // ---- one-time per-launch prep (deterministic) ----
  hipLaunchKernelGGL(init_state_kernel, dim3((2*BB*HH + 255)/256), dim3(256), 0, stream,
                     h0, c0, prev_out, h_state, c_state, outf32);
  hipLaunchKernelGGL(pack_weights_kernel, dim3(GG/16, KT0), dim3(32), 0, stream,
                     Wih0, EE + HH + 1, Whh0, HH, w0frag, KT0);
  hipLaunchKernelGGL(pack_weights_kernel, dim3(GG/16, KT1), dim3(32), 0, stream,
                     Wih1, HH, Whh1, HH, w1frag, KT1);
  hipLaunchKernelGGL(pack_weights_kernel, dim3(VV/16, KTG), dim3(32), 0, stream,
                     Wg, HH, (const float*)nullptr, 0, wgfrag, KTG);

  // ---- sequential scan over T steps ----
  for (int t = 0; t < TT; ++t) {
    hipLaunchKernelGGL(build_xh0_kernel, dim3((BB*KP0 + 255)/256), dim3(256), 0, stream,
                       ids, cdown, word_emb, spec_emb, outf32, h_state, xh0, t);
    // layer 0 gates: [32 x 1568(bf16)] @ w0frag -> [32 x 2048]
    hipLaunchKernelGGL(wmma_gemm_bias_kernel, dim3(GG/16), dim3(32), 0, stream,
                       xh0, KP0, w0frag, KT0, bih0, bhh0, gates0, GG);
    hipLaunchKernelGGL(lstm_epilogue_kernel, dim3((BB*HH + 255)/256), dim3(256), 0, stream,
                       gates0, h_state, c_state, xh1, h_state + BB*HH,
                       (float*)nullptr, (unsigned short*)nullptr);
    // layer 1 gates: [32 x 1024] @ w1frag -> [32 x 2048]
    hipLaunchKernelGGL(wmma_gemm_bias_kernel, dim3(GG/16), dim3(32), 0, stream,
                       xh1, KP1, w1frag, KT1, bih1, bhh1, gates1, GG);
    hipLaunchKernelGGL(lstm_epilogue_kernel, dim3((BB*HH + 255)/256), dim3(256), 0, stream,
                       gates1, h_state + BB*HH, c_state + BB*HH,
                       (unsigned short*)nullptr, (const float*)nullptr, outf32, outbf);
    // generator: [32 x 512] @ wgfrag -> [32 x 32000]
    hipLaunchKernelGGL(wmma_gemm_bias_kernel, dim3(VV/16), dim3(32), 0, stream,
                       outbf, KPG, wgfrag, KTG, bg, (const float*)nullptr, logits, VV);
    hipLaunchKernelGGL(logsoftmax_kernel, dim3(BB), dim3(256), 0, stream,
                       logits, out_f + (size_t)t * BB * VV);
  }

  // ---- tail outputs: (hT, cT, out) after scores ----
  float* tail = out_f + (size_t)TT * BB * VV;
  hipLaunchKernelGGL(finalize_kernel, dim3((2*2*BB*HH + BB*HH + 255)/256), dim3(256), 0, stream,
                     h_state, c_state, outf32, tail);
}
